// AttentionHead_85607288144432
// MI455X (gfx1250) — matrix operand order repaired and checked by reading, not compiled, not run on hardware
//
#include <hip/hip_runtime.h>
#include <hip/hip_bf16.h>

typedef __attribute__((ext_vector_type(16))) _Float16 v16h;
typedef __attribute__((ext_vector_type(8)))  float    v8f;
typedef __attribute__((ext_vector_type(4)))  int      v4i;
typedef _Float16 h16;

union V16HU { v16h v; uint4 q[2]; };

#define BATCH 8
#define SEQ   4096
#define DIM   128
#define KVT   64      // KV rows per tile
#define WAVES 8       // waves per attention block

#if __has_builtin(__builtin_amdgcn_global_load_async_to_lds_b128) && \
    __has_builtin(__builtin_amdgcn_s_wait_asynccnt)
#define USE_ASYNC_LDS 1
#else
#define USE_ASYNC_LDS 0
#endif

#if USE_ASYNC_LDS
__device__ inline __attribute__((address_space(1))) v4i* to_global_as(const void* p) {
    return (__attribute__((address_space(1))) v4i*)(uintptr_t)p;
}
__device__ inline __attribute__((address_space(3))) v4i* to_lds_as(void* p) {
    return (__attribute__((address_space(3))) v4i*)(uint32_t)(uintptr_t)p;
}
#endif

// ---------------------------------------------------------------------------
// Kernel 0: W (f32, [n][k]) -> W^T (f16, [k][n]) so WMMA B-fragments are
// contiguous 32-byte loads.
// ---------------------------------------------------------------------------
__global__ __launch_bounds__(256)
void wcvt_kernel(const float* __restrict__ Wq, const float* __restrict__ Wk,
                 const float* __restrict__ Wv,
                 h16* __restrict__ wqT, h16* __restrict__ wkT, h16* __restrict__ wvT) {
    const float* W = (blockIdx.x == 0) ? Wq : (blockIdx.x == 1) ? Wk : Wv;
    h16* T         = (blockIdx.x == 0) ? wqT : (blockIdx.x == 1) ? wkT : wvT;
    for (int idx = threadIdx.x; idx < DIM * DIM; idx += blockDim.x) {
        int n = idx >> 7;
        int k = idx & 127;
        T[k * DIM + n] = (h16)W[n * DIM + k];
    }
}

// ---------------------------------------------------------------------------
// Kernel 1: q/k/v projection (x @ W^T) with WMMA f16, fused LayerNorm on q,k.
// One block = 3 waves; wave 0 -> Q(+LN, *1/sqrt(D)), wave 1 -> K(+LN), wave 2 -> V.
// ---------------------------------------------------------------------------
__global__ __launch_bounds__(96)
void qkv_proj_ln_kernel(const float* __restrict__ x,
                        const h16* __restrict__ wqT, const h16* __restrict__ wkT,
                        const h16* __restrict__ wvT,
                        const float* __restrict__ qn_w, const float* __restrict__ qn_b,
                        const float* __restrict__ kn_w, const float* __restrict__ kn_b,
                        h16* __restrict__ qh, h16* __restrict__ kh, h16* __restrict__ vh) {
    const int wave = threadIdx.x >> 5;
    const int lane = threadIdx.x & 31;
    const int m    = lane & 15;
    const int half = lane >> 4;
    const int rowbase = blockIdx.x * 16;

    const h16* wT = (wave == 0) ? wqT : (wave == 1) ? wkT : wvT;
    h16* outp     = (wave == 0) ? qh  : (wave == 1) ? kh  : vh;

    // A fragments of x (ISA 16-bit A layout), f32 -> f16
    V16HU a[4];
    const float* xrow = x + (size_t)(rowbase + m) * DIM;
#pragma unroll
    for (int ks = 0; ks < 4; ++ks) {
        const float* p0 = xrow + ks * 32 + 8 * half;
        const float* p1 = p0 + 16;
        v16h t;
#pragma unroll
        for (int i = 0; i < 8; ++i) { t[i] = (h16)p0[i]; t[8 + i] = (h16)p1[i]; }
        a[ks].v = t;
    }

    v8f c[8];
#pragma unroll
    for (int nt = 0; nt < 8; ++nt) {
        v8f acc = {};
#pragma unroll
        for (int ks = 0; ks < 4; ++ks) {
            V16HU b;
            for (int i = 0; i < 8; ++i) {
                b.v[i]     = wT[(size_t)(ks * 32 + 8 * half + i) * DIM + nt * 16 + m];
                b.v[8 + i] = wT[(size_t)(ks * 32 + 16 + 8 * half + i) * DIM + nt * 16 + m];
            }
            acc = __builtin_amdgcn_wmma_f32_16x16x32_f16(false, a[ks].v, false, b.v,
                                                         (short)0, acc, false, false);
        }
        c[nt] = acc;
    }

    if (wave < 2) {
        const float* nw = (wave == 0) ? qn_w : kn_w;
        const float* nb = (wave == 0) ? qn_b : kn_b;
        float psum[8], psq[8];
#pragma unroll
        for (int i = 0; i < 8; ++i) { psum[i] = 0.f; psq[i] = 0.f; }
#pragma unroll
        for (int nt = 0; nt < 8; ++nt)
#pragma unroll
            for (int i = 0; i < 8; ++i) {
                float v = c[nt][i];
                psum[i] += v;
                psq[i]  += v * v;
            }
#pragma unroll
        for (int off = 1; off < 16; off <<= 1)
#pragma unroll
            for (int i = 0; i < 8; ++i) {
                psum[i] += __shfl_xor(psum[i], off, 32);
                psq[i]  += __shfl_xor(psq[i],  off, 32);
            }
        const float invD = 1.0f / 128.0f;
        float mu[8], rstd[8];
#pragma unroll
        for (int i = 0; i < 8; ++i) {
            mu[i] = psum[i] * invD;
            float var = psq[i] * invD - mu[i] * mu[i];
            rstd[i] = rsqrtf(var + 1e-5f);
        }
        const float qscale = (wave == 0) ? 0.08838834764831845f : 1.0f; // 1/sqrt(D)
#pragma unroll
        for (int nt = 0; nt < 8; ++nt) {
            int col = nt * 16 + m;
            float w  = nw[col];
            float bb = nb[col];
#pragma unroll
            for (int i = 0; i < 8; ++i) {
                float v = ((c[nt][i] - mu[i]) * rstd[i] * w + bb) * qscale;
                int row = rowbase + i + 8 * half;
                outp[(size_t)row * DIM + col] = (h16)v;
            }
        }
    } else {
#pragma unroll
        for (int nt = 0; nt < 8; ++nt) {
            int col = nt * 16 + m;
#pragma unroll
            for (int i = 0; i < 8; ++i) {
                int row = rowbase + i + 8 * half;
                outp[(size_t)row * DIM + col] = (h16)c[nt][i];
            }
        }
    }
}

// ---------------------------------------------------------------------------
// Kernel 2: flash attention, software-pipelined.
//  - 8 waves x 16 Q rows; KV tile = 64 rows shared via LDS.
//  - K staged through registers (double-buffered), stored transposed to LDS.
//  - V staged with gfx1250 async global->LDS loads (ping-pong buffer) when the
//    builtin is available; register path otherwise.
//  - Per tile per wave: 16 WMMAs (Q@K^T) + 16 WMMAs (P@V), f32_16x16x32_f16.
// ---------------------------------------------------------------------------
__global__ __launch_bounds__(256)
void flash_attn_kernel(const h16* __restrict__ qh, const h16* __restrict__ kh,
                       const h16* __restrict__ vh, float* __restrict__ out) {
    __shared__ __align__(16) h16 kT[DIM * KVT];              // K^T: [feat][kv_row] 16 KB
#if USE_ASYNC_LDS
    __shared__ __align__(16) h16 vt[2][KVT * DIM];           // V ping-pong        32 KB
#else
    __shared__ __align__(16) h16 vt[1][KVT * DIM];           // V                  16 KB
#endif
    __shared__ __align__(16) h16 pbuf[WAVES][16 * KVT];      // wave-private P     16 KB

    const int tid  = threadIdx.x;
    const int wave = tid >> 5;
    const int lane = tid & 31;
    const int m    = lane & 15;
    const int half = lane >> 4;
    const int b    = blockIdx.y;
    const int qbase = blockIdx.x * (WAVES * 16) + wave * 16;
    const size_t boff = (size_t)b * SEQ * DIM;

    // Q A-fragments held in registers for the whole pass
    V16HU qa[4];
    const h16* qrow = qh + boff + (size_t)(qbase + m) * DIM;
#pragma unroll
    for (int ks = 0; ks < 4; ++ks) {
        qa[ks].q[0] = *(const uint4*)(qrow + ks * 32 + 8 * half);
        qa[ks].q[1] = *(const uint4*)(qrow + ks * 32 + 16 + 8 * half);
    }

    v8f o[8];
#pragma unroll
    for (int nt = 0; nt < 8; ++nt) o[nt] = (v8f){};
    float mstate[8], lstate[8];
#pragma unroll
    for (int i = 0; i < 8; ++i) { mstate[i] = -1.0e30f; lstate[i] = 0.f; }

    const h16* kbase = kh + boff;
    const h16* vbase = vh + boff;

    // staging assignment: 64 B per thread per matrix per tile
    const int ldr = tid >> 2;          // 0..63  kv row
    const int ldf = (tid & 3) * 32;    // 0..96  feature base

    uint4 kreg[4];
#if !USE_ASYNC_LDS
    uint4 vreg[4];
#endif

    // ---- prologue: start tile 0
    {
        const uint4* ksrc = (const uint4*)(kbase + (size_t)ldr * DIM + ldf);
#pragma unroll
        for (int w = 0; w < 4; ++w) kreg[w] = ksrc[w];
#if USE_ASYNC_LDS
        const h16* vsrc = vbase + (size_t)ldr * DIM + ldf;
        h16* vdst = &vt[0][ldr * DIM + ldf];
#pragma unroll
        for (int w = 0; w < 4; ++w)
            __builtin_amdgcn_global_load_async_to_lds_b128(
                to_global_as(vsrc + 8 * w), to_lds_as(vdst + 8 * w), 0, 0);
#else
        const uint4* vsrc = (const uint4*)(vbase + (size_t)ldr * DIM + ldf);
#pragma unroll
        for (int w = 0; w < 4; ++w) vreg[w] = vsrc[w];
#endif
    }

    for (int t = 0; t < SEQ / KVT; ++t) {
#if USE_ASYNC_LDS
        const int cur = t & 1;
#else
        const int cur = 0;
#endif
        __syncthreads();   // previous tile's compute finished reading LDS

        // ---- commit staged K (transposed) into LDS
        {
            union { uint4 q[4]; h16 h[32]; } ku;
#pragma unroll
            for (int w = 0; w < 4; ++w) ku.q[w] = kreg[w];
#pragma unroll
            for (int i = 0; i < 32; ++i)
                kT[(ldf + i) * KVT + ldr] = ku.h[i];
        }
#if USE_ASYNC_LDS
        __builtin_amdgcn_s_wait_asynccnt(0);   // V tile t landed in vt[cur]
#else
        {
            uint4* vdst = (uint4*)(&vt[0][ldr * DIM + ldf]);
#pragma unroll
            for (int w = 0; w < 4; ++w) vdst[w] = vreg[w];
        }
#endif
        __syncthreads();

        // ---- start fetching tile t+1 while we compute tile t
        if (t + 1 < SEQ / KVT) {
            const size_t nb = (size_t)((t + 1) * KVT + ldr) * DIM + ldf;
            const uint4* ksrc = (const uint4*)(kbase + nb);
#pragma unroll
            for (int w = 0; w < 4; ++w) kreg[w] = ksrc[w];
#if USE_ASYNC_LDS
            const h16* vsrc = vbase + nb;
            h16* vdst = &vt[1 - cur][ldr * DIM + ldf];
#pragma unroll
            for (int w = 0; w < 4; ++w)
                __builtin_amdgcn_global_load_async_to_lds_b128(
                    to_global_as(vsrc + 8 * w), to_lds_as(vdst + 8 * w), 0, 0);
#else
            const uint4* vsrc = (const uint4*)(vbase + nb);
#pragma unroll
            for (int w = 0; w < 4; ++w) vreg[w] = vsrc[w];
#endif
        }

        // ---- S = Q @ K^T : four 16-col chunks, 16 WMMAs
        v8f sc[4];
#pragma unroll
        for (int jc = 0; jc < 4; ++jc) {
            v8f acc = {};
#pragma unroll
            for (int ks = 0; ks < 4; ++ks) {
                V16HU bf;   // lane = kv col, elements = k-features (table order)
                for (int i = 0; i < 8; ++i) {
                    bf.v[i]     = kT[(ks * 32 + 8 * half + i) * KVT + jc * 16 + m];
                    bf.v[8 + i] = kT[(ks * 32 + 16 + 8 * half + i) * KVT + jc * 16 + m];
                }
                acc = __builtin_amdgcn_wmma_f32_16x16x32_f16(false, qa[ks].v, false, bf.v,
                                                             (short)0, acc, false, false);
            }
            sc[jc] = acc;
        }

        // ---- online softmax over 64 columns
        float mr[8];
#pragma unroll
        for (int i = 0; i < 8; ++i)
            mr[i] = fmaxf(fmaxf(sc[0][i], sc[1][i]), fmaxf(sc[2][i], sc[3][i]));
#pragma unroll
        for (int off = 1; off < 16; off <<= 1)
#pragma unroll
            for (int i = 0; i < 8; ++i) mr[i] = fmaxf(mr[i], __shfl_xor(mr[i], off, 32));

        float alpha[8], ps[8];
#pragma unroll
        for (int i = 0; i < 8; ++i) {
            float mn = fmaxf(mstate[i], mr[i]);
            alpha[i] = __expf(mstate[i] - mn);
            mstate[i] = mn;
            ps[i] = 0.f;
        }
#pragma unroll
        for (int jc = 0; jc < 4; ++jc)
#pragma unroll
            for (int i = 0; i < 8; ++i) {
                float p = __expf(sc[jc][i] - mstate[i]);
                sc[jc][i] = p;
                ps[i] += p;
            }
#pragma unroll
        for (int off = 1; off < 16; off <<= 1)
#pragma unroll
            for (int i = 0; i < 8; ++i) ps[i] += __shfl_xor(ps[i], off, 32);
#pragma unroll
        for (int i = 0; i < 8; ++i) lstate[i] = lstate[i] * alpha[i] + ps[i];

#pragma unroll
        for (int nt = 0; nt < 8; ++nt)
#pragma unroll
            for (int i = 0; i < 8; ++i) o[nt][i] = o[nt][i] * alpha[i];

        // ---- C-layout -> A-layout for P via wave-private LDS (in-order per wave)
        h16* pw = &pbuf[wave][0];
#pragma unroll
        for (int jc = 0; jc < 4; ++jc)
#pragma unroll
            for (int i = 0; i < 8; ++i)
                pw[(i + 8 * half) * KVT + jc * 16 + m] = (h16)sc[jc][i];
        asm volatile("" ::: "memory");

        V16HU pa[2];
#pragma unroll
        for (int kc = 0; kc < 2; ++kc) {
            pa[kc].q[0] = *(const uint4*)(pw + m * KVT + kc * 32 + 8 * half);
            pa[kc].q[1] = *(const uint4*)(pw + m * KVT + kc * 32 + 16 + 8 * half);
        }

        // ---- O += P @ V : 16 WMMAs
        const h16* vcur = &vt[cur][0];
#pragma unroll
        for (int nt = 0; nt < 8; ++nt) {
#pragma unroll
            for (int kc = 0; kc < 2; ++kc) {
                V16HU vb;   // lane = feature col, elements = kv rows (table order)
                for (int i = 0; i < 8; ++i) {
                    vb.v[i]     = vcur[(kc * 32 + 8 * half + i) * DIM + nt * 16 + m];
                    vb.v[8 + i] = vcur[(kc * 32 + 16 + 8 * half + i) * DIM + nt * 16 + m];
                }
                o[nt] = __builtin_amdgcn_wmma_f32_16x16x32_f16(false, pa[kc].v, false, vb.v,
                                                               (short)0, o[nt], false, false);
            }
        }
    }

    // ---- epilogue: normalize by running sum, store fp32
    float rl[8];
#pragma unroll
    for (int i = 0; i < 8; ++i) rl[i] = 1.0f / lstate[i];
    float* obase = out + boff;
#pragma unroll
    for (int nt = 0; nt < 8; ++nt) {
        int col = nt * 16 + m;
#pragma unroll
        for (int i = 0; i < 8; ++i) {
            int row = qbase + i + 8 * half;
            obase[(size_t)row * DIM + col] = o[nt][i] * rl[i];
        }
    }
}

// ---------------------------------------------------------------------------
extern "C" void kernel_launch(void* const* d_in, const int* in_sizes, int n_in,
                              void* d_out, int out_size, void* d_ws, size_t ws_size,
                              hipStream_t stream) {
    const float* x   = (const float*)d_in[0];
    const float* Wq  = (const float*)d_in[1];
    const float* Wk  = (const float*)d_in[2];
    const float* Wv  = (const float*)d_in[3];
    const float* qnw = (const float*)d_in[4];
    const float* qnb = (const float*)d_in[5];
    const float* knw = (const float*)d_in[6];
    const float* knb = (const float*)d_in[7];
    float* out = (float*)d_out;

    const size_t nq = (size_t)BATCH * SEQ * DIM;
    h16* qh  = (h16*)d_ws;
    h16* kh  = qh + nq;
    h16* vh  = kh + nq;
    h16* wqT = vh + nq;
    h16* wkT = wqT + DIM * DIM;
    h16* wvT = wkT + DIM * DIM;

    wcvt_kernel<<<3, 256, 0, stream>>>(Wq, Wk, Wv, wqT, wkT, wvT);
    qkv_proj_ln_kernel<<<(BATCH * SEQ) / 16, 96, 0, stream>>>(
        x, wqT, wkT, wvT, qnw, qnb, knw, knb, qh, kh, vh);
    flash_attn_kernel<<<dim3(SEQ / (WAVES * 16), BATCH), 256, 0, stream>>>(
        qh, kh, vh, out);
}
